// GCNEncoder_22265110463394
// MI455X (gfx1250) — matrix operand order repaired and checked by reading, not compiled, not run on hardware
//
#include <hip/hip_runtime.h>

#define B_     4
#define N_     2048
#define F_IN_  32
#define DIM_   64
#define HEADS_ 4
#define DH_    16
#define DEPTH_ 3
#define SCALE_ 0.25f
#define NEG_   -1e30f

typedef _Float16 v8h  __attribute__((ext_vector_type(8)));
typedef _Float16 v16h __attribute__((ext_vector_type(16)));
typedef float    v8f  __attribute__((ext_vector_type(8)));

__device__ __forceinline__ v8f wmma_f16(v16h a, v16h b, v8f c) {
  return __builtin_amdgcn_wmma_f32_16x16x32_f16(false, a, false, b,
                                                (short)0, c, false, false);
}

// ---------------------------------------------------------------------------
// K1: x = relu(nf @ W_fc + b_fc)   one row per block, 64 threads (one per col)
// ---------------------------------------------------------------------------
__global__ void fc_relu_kernel(const float* __restrict__ nf,
                               const float* __restrict__ Wfc,
                               const float* __restrict__ bfc,
                               float* __restrict__ x) {
  __shared__ float sh[F_IN_];
  const int row = blockIdx.x;
  const int t   = threadIdx.x;
  if (t < F_IN_) sh[t] = nf[(size_t)row * F_IN_ + t];
  __syncthreads();
  float acc = bfc[t];
#pragma unroll
  for (int d = 0; d < F_IN_; ++d) acc = fmaf(sh[d], Wfc[d * DIM_ + t], acc);
  x[(size_t)row * DIM_ + t] = fmaxf(acc, 0.f);
}

// ---------------------------------------------------------------------------
// K2: h = layernorm(x); q,k,v = h@W + b, repacked to f16 in WMMA-ready layouts:
//   q  : [B][H][N][16]                (B operand rows, contiguous DH)
//   khp: [B][H][N][32] padded rows    [dh0..7 | 0 x8 | dh8..15 | 0 x8]
//        -> A operand is ONE aligned v16h load at +hi*16
//   vT : [B][H][DH][N] with keys bit3<->bit4 swapped inside each 32-group
//        ([k0..7,k16..23,k8..15,k24..31]) -> V^T A operand is one v16h load
// ---------------------------------------------------------------------------
__global__ void ln_qkv_kernel(const float* __restrict__ x,
                              const float* __restrict__ ln_g,
                              const float* __restrict__ ln_b,
                              const float* __restrict__ Wq,
                              const float* __restrict__ bq,
                              const float* __restrict__ Wkv,
                              const float* __restrict__ bkv,
                              _Float16* __restrict__ qh,
                              _Float16* __restrict__ khp,
                              _Float16* __restrict__ vT) {
  __shared__ float shx[DIM_];
  __shared__ float shh[DIM_];
  const int row = blockIdx.x;            // 0..B*N-1
  const int b   = row / N_;
  const int n   = row % N_;
  const int t   = threadIdx.x;           // 0..63
  const float v = x[(size_t)row * DIM_ + t];
  shx[t] = v;
  __syncthreads();
  float mu = 0.f;
#pragma unroll
  for (int d = 0; d < DIM_; ++d) mu += shx[d];
  mu *= (1.f / DIM_);
  float var = 0.f;
#pragma unroll
  for (int d = 0; d < DIM_; ++d) { float c = shx[d] - mu; var = fmaf(c, c, var); }
  var *= (1.f / DIM_);
  const float hval = (v - mu) * rsqrtf(var + 1e-5f) * ln_g[t] + ln_b[t];
  shh[t] = hval;
  __syncthreads();
  float q = bq[t], k = bkv[t], vv = bkv[DIM_ + t];
#pragma unroll
  for (int d = 0; d < DIM_; ++d) {
    const float hd = shh[d];
    q  = fmaf(hd, Wq[d * DIM_ + t], q);
    k  = fmaf(hd, Wkv[d * 2 * DIM_ + t], k);
    vv = fmaf(hd, Wkv[d * 2 * DIM_ + DIM_ + t], vv);
  }
  const int hh = t >> 4, dd = t & 15;
  const size_t bh = (size_t)(b * HEADS_ + hh);
  qh[(bh * N_ + n) * DH_ + dd] = (_Float16)q;
  // padded K row: dd<8 -> slot dd ; dd>=8 -> slot dd+8 ; mirror slot+8 zeroed
  {
    _Float16* krow = khp + (bh * N_ + n) * 32;
    const int kpos = dd + (dd & 8);
    krow[kpos]     = (_Float16)k;
    krow[kpos + 8] = (_Float16)0.f;
  }
  // swizzled V^T: swap bit3<->bit4 of key index inside each 32-group
  {
    const int nsw = (n & ~0x18) | ((n & 8) << 1) | ((n & 16) >> 1);
    vT[(bh * DH_ + dd) * N_ + nsw] = (_Float16)vv;
  }
}

// ---------------------------------------------------------------------------
// K3: flash attention, transposed tiles: sim^T = K x Q^T, O^T = V^T x P^T.
//   One block = 4 waves = 4 heads for the same (b, 16-query tile); the
//   adjacency tile is staged ONCE to LDS as an additive bias (0 / -1e30),
//   software-pipelined one 32-key tile ahead so its HBM latency hides
//   behind the current tile's WMMAs (in-order VMEM completion means the
//   K-tile wait covers it for free).
//   Queries live in the lane (N) dim => per-lane scalar softmax stats,
//   key reduction = 8-VGPR VALU reduce + one shfl_xor(16).
//   No exp guards: exp(-1e30 - m) underflows to 0, and the all-masked
//   transient self-corrects via alpha = exp(m_old - m_new) = 0 (every
//   query has a self-loop, so final stats are always clean).
// ---------------------------------------------------------------------------
__global__ void attn_kernel(const _Float16* __restrict__ qh,
                            const _Float16* __restrict__ khp,
                            const _Float16* __restrict__ vT,
                            const float* __restrict__ adj,
                            float* __restrict__ o_inner) {
  __shared__ __align__(64) float    adjS[16][32];            // bias [query][key]
  __shared__ __align__(64) _Float16 Pt[HEADS_][16][32];      // P^T f16
  const int bid  = blockIdx.x;          // B*(N/16) = 512
  const int qt   = bid & 127;
  const int b    = bid >> 7;
  const int q0   = qt * 16;
  const int tid  = threadIdx.x;         // 0..127
  const int h    = tid >> 5;            // wave id = head
  const int lane = tid & 31;
  const int hi   = lane >> 4;           // lane half
  const int l16  = lane & 15;
  const int kb   = hi ? 8 : 0;

  const size_t bh = (size_t)(b * HEADS_ + h);
  const _Float16* kbase = khp + bh * N_ * 32;
  const _Float16* vrow  = vT + (bh * DH_ + l16) * N_;

  // B operand (loop-invariant): q^T (dh x 16 queries); elems 0-7 = dh hi*8..hi*8+7, elems 8-15 are K-pad.
  v16h bqv = {};
  {
    const v8h qv = *(const v8h*)(qh + (bh * N_ + q0 + l16) * DH_ + hi * 8);
#pragma unroll
    for (int s = 0; s < 8; ++s) bqv[s] = qv[s];
  }

  float m_run = NEG_, l_run = 0.f;
  v8f acc = {};
  const v8f zc = {};

  const int srow = tid >> 3;            // 0..15 adj staging row
  const int sseg = (tid & 7) * 4;       // 0..28 adj staging col
  const float* adjp = adj + ((size_t)b * N_ + q0 + srow) * N_ + sseg;

  float4 nxt = *(const float4*)adjp;    // pipelined adj tile (j = 0)

  for (int j = 0; j < N_; j += 32) {
    // publish current adj tile as additive bias (0 / -1e30)
    {
      float4 cur = nxt;
      cur.x = (cur.x > 0.f) ? 0.f : NEG_;
      cur.y = (cur.y > 0.f) ? 0.f : NEG_;
      cur.z = (cur.z > 0.f) ? 0.f : NEG_;
      cur.w = (cur.w > 0.f) ? 0.f : NEG_;
      *(float4*)&adjS[srow][sseg] = cur;
    }
    __syncthreads();
    // issue next tile's load now; consumed at the top of the next iteration
    if (j + 32 < N_) nxt = *(const float4*)(adjp + j + 32);

    // A tiles: single aligned v16h loads (padded layout supplies the K-pad)
    const v16h ak0 = *(const v16h*)(kbase + (size_t)(j + l16) * 32 + hi * 16);
    const v16h ak1 = *(const v16h*)(kbase + (size_t)(j + 16 + l16) * 32 + hi * 16);

    v8f c0 = wmma_f16(ak0, bqv, zc);    // sim^T[key = j+kb+r][query = l16]
    v8f c1 = wmma_f16(ak1, bqv, zc);    // sim^T[key = j+16+kb+r][query]

    // additive mask bias from LDS (contiguous in key index)
    const float4 a0a = *(const float4*)&adjS[l16][kb];
    const float4 a0b = *(const float4*)&adjS[l16][kb + 4];
    const float4 a1a = *(const float4*)&adjS[l16][16 + kb];
    const float4 a1b = *(const float4*)&adjS[l16][16 + kb + 4];
    const float a0[8] = {a0a.x, a0a.y, a0a.z, a0a.w, a0b.x, a0b.y, a0b.z, a0b.w};
    const float a1[8] = {a1a.x, a1a.y, a1a.z, a1a.w, a1b.x, a1b.y, a1b.z, a1b.w};

    float s0[8], s1[8];
    float mx = NEG_;
#pragma unroll
    for (int r = 0; r < 8; ++r) {
      s0[r] = fmaf(c0[r], SCALE_, a0[r]);
      s1[r] = fmaf(c1[r], SCALE_, a1[r]);
      mx = fmaxf(mx, fmaxf(s0[r], s1[r]));
    }
    mx = fmaxf(mx, __shfl_xor(mx, 16, 32));    // combine the two key-halves
    const float mnew = fmaxf(m_run, mx);
    const float al   = __expf(m_run - mnew);

    float rs = 0.f;
    v8h p0, p1;
#pragma unroll
    for (int r = 0; r < 8; ++r) {
      const float e0 = __expf(s0[r] - mnew);
      const float e1 = __expf(s1[r] - mnew);
      rs += e0 + e1;
      p0[r] = (_Float16)e0;
      p1[r] = (_Float16)e1;
    }
    rs += __shfl_xor(rs, 16, 32);
    l_run = l_run * al + rs;
    m_run = mnew;

    // P^T through LDS: each lane owns 8+8 contiguous key slots of its query row
    *(v8h*)&Pt[h][l16][hi * 16]     = p0;
    *(v8h*)&Pt[h][l16][hi * 16 + 8] = p1;
    __syncthreads();   // also fences adjS reads vs next-iter restage
    // B operand P^T: lanes 0-15 K=keys 0..7,16..23, lanes 16-31 K=keys 8..15,24..31
    const v16h bp = *(const v16h*)&Pt[h][l16][hi * 16];

    // A operand V^T: swizzled layout -> one aligned v16h load
    const v16h av = *(const v16h*)(vrow + j + hi * 16);

#pragma unroll
    for (int r = 0; r < 8; ++r) acc[r] *= al;  // per-lane scalar rescale
    acc = wmma_f16(av, bp, acc);
  }

  // O^T element (r,lane): dh = kb + r (contiguous 8), query = l16
  const float inv = (l_run > 0.f) ? 1.f / l_run : 0.f;
  float* orow = o_inner + ((size_t)b * N_ + q0 + l16) * DIM_ + h * DH_ + kb;
  float4 oA, oB;
  oA.x = acc[0] * inv; oA.y = acc[1] * inv; oA.z = acc[2] * inv; oA.w = acc[3] * inv;
  oB.x = acc[4] * inv; oB.y = acc[5] * inv; oB.z = acc[6] * inv; oB.w = acc[7] * inv;
  *(float4*)orow       = oA;
  *(float4*)(orow + 4) = oB;
}

// ---------------------------------------------------------------------------
// K4: o = o_inner @ Wo + bo; gate = sigmoid([o,x,o-x] @ Wg); x = lerp
// ---------------------------------------------------------------------------
__global__ void out_gate_kernel(const float* __restrict__ o_inner,
                                const float* __restrict__ Wo,
                                const float* __restrict__ bo,
                                const float* __restrict__ Wg,
                                const float* __restrict__ xin,
                                float* __restrict__ xout) {
  __shared__ float sho[DIM_];
  __shared__ float red[DIM_];
  const int row = blockIdx.x;
  const int t   = threadIdx.x;
  sho[t] = o_inner[(size_t)row * DIM_ + t];
  __syncthreads();
  float o = bo[t];
#pragma unroll
  for (int d = 0; d < DIM_; ++d) o = fmaf(sho[d], Wo[d * DIM_ + t], o);
  const float xv = xin[(size_t)row * DIM_ + t];
  red[t] = o * Wg[t] + xv * Wg[DIM_ + t] + (o - xv) * Wg[2 * DIM_ + t];
  __syncthreads();
  float s = 0.f;
#pragma unroll
  for (int d = 0; d < DIM_; ++d) s += red[d];
  const float g = 1.f / (1.f + __expf(-s));
  xout[(size_t)row * DIM_ + t] = o * g + xv * (1.f - g);
}

// ---------------------------------------------------------------------------
extern "C" void kernel_launch(void* const* d_in, const int* in_sizes, int n_in,
                              void* d_out, int out_size, void* d_ws, size_t ws_size,
                              hipStream_t stream) {
  (void)in_sizes; (void)n_in; (void)out_size; (void)ws_size;
  const float* nf  = (const float*)d_in[0];
  const float* adj = (const float*)d_in[1];
  const float* Wfc = (const float*)d_in[2];
  const float* bfc = (const float*)d_in[3];
  const float* lng = (const float*)d_in[4];
  const float* lnb = (const float*)d_in[5];
  const float* Wq  = (const float*)d_in[6];
  const float* bq  = (const float*)d_in[7];
  const float* Wkv = (const float*)d_in[8];
  const float* bkv = (const float*)d_in[9];
  const float* Wo  = (const float*)d_in[10];
  const float* bo  = (const float*)d_in[11];
  const float* Wg  = (const float*)d_in[12];

  char* ws = (char*)d_ws;
  float*    xbuf    = (float*)ws;                                     // 2 MB
  float*    o_inner = (float*)(ws + (size_t)2 * 1024 * 1024);         // 2 MB
  _Float16* qh      = (_Float16*)(ws + (size_t)4 * 1024 * 1024);      // 512 KB
  _Float16* khp     = (_Float16*)(ws + (size_t)4 * 1024 * 1024 + 512 * 1024); // 1 MB
  _Float16* vT      = (_Float16*)(ws + (size_t)5 * 1024 * 1024 + 512 * 1024); // 512 KB

  const int rows = B_ * N_;
  fc_relu_kernel<<<rows, DIM_, 0, stream>>>(nf, Wfc, bfc, xbuf);
  for (int dpt = 0; dpt < DEPTH_; ++dpt) {
    ln_qkv_kernel<<<rows, DIM_, 0, stream>>>(
        xbuf, lng + dpt * DIM_, lnb + dpt * DIM_,
        Wq + (size_t)dpt * DIM_ * DIM_, bq + dpt * DIM_,
        Wkv + (size_t)dpt * DIM_ * 2 * DIM_, bkv + dpt * 2 * DIM_,
        qh, khp, vT);
    attn_kernel<<<B_ * (N_ / 16), 4 * 32, 0, stream>>>(qh, khp, vT, adj, o_inner);
    out_gate_kernel<<<rows, DIM_, 0, stream>>>(
        o_inner, Wo + (size_t)dpt * DIM_ * DIM_, bo + dpt * DIM_,
        Wg + (size_t)dpt * 3 * DIM_, xbuf,
        (dpt == DEPTH_ - 1) ? (float*)d_out : xbuf);
  }
}
